// KernelAttention_9947144258099
// MI455X (gfx1250) — matrix operand order repaired and checked by reading, not compiled, not run on hardware
//
#include <hip/hip_runtime.h>
#include <hip/hip_bf16.h>

typedef __attribute__((ext_vector_type(16))) _Float16 v16h;
typedef __attribute__((ext_vector_type(8)))  _Float16 v8h;
typedef __attribute__((ext_vector_type(8)))  float    v8f;

#define BATCH   4
#define HEADS   16
#define SEQ     4096
#define DIM     64
#define NBH     (BATCH * HEADS)
#define NCHUNK  8
#define CHUNK_S (SEQ / NCHUNK)      // 512 rows of S per pass-1 workgroup
#define KSTEP   32
#define NKSTEP  (CHUNK_S / KSTEP)   // 16
#define LDS_STRIDE 40               // halves per LDS row (80B): keeps b128 reads 16B-aligned
#define PART_STRIDE (DIM * DIM + DIM)   // 4160 f32 per (bh,chunk): KV[64][64] + k_one[64]
#define KVT_ROWS (DIM + 1)          // rows e=0..63: KV^T ; row 64: k_one
#define PART_BYTES ((size_t)NBH * NCHUNK * PART_STRIDE * sizeof(float))
#define KVT_BYTES  ((size_t)NBH * KVT_ROWS * DIM * sizeof(_Float16))

__device__ __forceinline__ float phi_elu1(float x) {
    // elu(x)+1 with alpha=1:  x>0 -> x+1 ; else exp(x)
    return x > 0.0f ? x + 1.0f : __expf(x);
}

__device__ __forceinline__ v8f wmma16(v16h a, v16h b, v8f c) {
    return __builtin_amdgcn_wmma_f32_16x16x32_f16(false, a, false, b, (short)0, c,
                                                  false, false);
}

// ---------------------------------------------------------------------------
// Pass 1: per (bh, chunk) accumulate KV[d,e] = sum_s phi_k[s,d] * V[s,e] and
// k_one[d] = sum_s phi_k[s,d] (via an all-ones augmented B column tile).
// ---------------------------------------------------------------------------
__global__ __launch_bounds__(128)
void la_pass1_kv(const float* __restrict__ K, const float* __restrict__ V,
                 const float* __restrict__ mask, float* __restrict__ partial) {
    __shared__ _Float16 KT[DIM * LDS_STRIDE];   // phi_k transposed: [d][s]
    __shared__ _Float16 VT[DIM * LDS_STRIDE];   // V     transposed: [e][s]

    const int bh    = blockIdx.x >> 3;          // NCHUNK == 8
    const int chunk = blockIdx.x & 7;
    const int b     = bh >> 4;                  // HEADS == 16
    const int tid   = threadIdx.x;
    const int lane  = tid & 31;
    const int wave  = tid >> 5;                 // wave owns d-rows [wave*16, wave*16+16)
    const int half  = (lane >> 4) & 1;
    const int l15   = lane & 15;

    const float scale = 0.35355339059327379f;   // 64^-0.25
    const size_t base = ((size_t)bh * SEQ + (size_t)chunk * CHUNK_S) * DIM;
    const float* kp = K + base;
    const float* vp = V + base;
    const float* mp = mask + (size_t)b * SEQ + (size_t)chunk * CHUNK_S;

    v8f acc0 = {}, acc1 = {}, acc2 = {}, acc3 = {}, acc4 = {};

    // augmented ones tile: B[k][n] = (n==0) ? 1 : 0  ->  D[:,0] = row-sums of A
    v16h bones;
    {
        const _Float16 o = (l15 == 0) ? (_Float16)1.0f : (_Float16)0.0f;
#pragma unroll
        for (int i = 0; i < 16; ++i) bones[i] = o;
    }

    const int d_row  = wave * 16 + l15;
    const int a_off0 = d_row * LDS_STRIDE + half * 8;   // ISA A layout: half picks K 8-blocks

    for (int ks = 0; ks < NKSTEP; ++ks) {
        const int s0 = ks * KSTEP;
        __syncthreads();
        // Stage 32(s) x 64(d) f32 tiles coalesced, write f16 transposed into LDS.
#pragma unroll
        for (int i = 0; i < 4; ++i) {
            const int s  = (tid >> 4) + i * 8;      // 0..31
            const int d4 = (tid & 15) * 4;
            const float4 kk = *(const float4*)(kp + (size_t)(s0 + s) * DIM + d4);
            const float4 vv = *(const float4*)(vp + (size_t)(s0 + s) * DIM + d4);
            const float  m  = mp[s0 + s];
            KT[(d4 + 0) * LDS_STRIDE + s] = (_Float16)(phi_elu1(kk.x * scale) * m);
            KT[(d4 + 1) * LDS_STRIDE + s] = (_Float16)(phi_elu1(kk.y * scale) * m);
            KT[(d4 + 2) * LDS_STRIDE + s] = (_Float16)(phi_elu1(kk.z * scale) * m);
            KT[(d4 + 3) * LDS_STRIDE + s] = (_Float16)(phi_elu1(kk.w * scale) * m);
            VT[(d4 + 0) * LDS_STRIDE + s] = (_Float16)vv.x;
            VT[(d4 + 1) * LDS_STRIDE + s] = (_Float16)vv.y;
            VT[(d4 + 2) * LDS_STRIDE + s] = (_Float16)vv.z;
            VT[(d4 + 3) * LDS_STRIDE + s] = (_Float16)vv.w;
        }
        __syncthreads();

        // A = phi_k^T tile (16 d x 32 s), per-lane K sets per ISA 16-bit A layout.
        const v8h alo = *(const v8h*)(&KT[a_off0]);
        const v8h ahi = *(const v8h*)(&KT[a_off0 + 16]);
        const v16h a = __builtin_shufflevector(alo, ahi,
            0,1,2,3,4,5,6,7,8,9,10,11,12,13,14,15);

        auto loadB = [&](int nt) -> v16h {
            const int idx = (nt * 16 + l15) * LDS_STRIDE + half * 8;
            const v8h lo = *(const v8h*)(&VT[idx]);
            const v8h hi = *(const v8h*)(&VT[idx + 16]);
            return __builtin_shufflevector(lo, hi,
                0,1,2,3,4,5,6,7,8,9,10,11,12,13,14,15);
        };

        // Materialize all B tiles in distinct registers so the ds_loads can
        // clause together and the five WMMAs issue back-to-back.
        const v16h b0t = loadB(0);
        const v16h b1t = loadB(1);
        const v16h b2t = loadB(2);
        const v16h b3t = loadB(3);

        acc0 = wmma16(a, b0t,   acc0);
        acc1 = wmma16(a, b1t,   acc1);
        acc2 = wmma16(a, b2t,   acc2);
        acc3 = wmma16(a, b3t,   acc3);
        acc4 = wmma16(a, bones, acc4);
    }

    // D layout: element (m = r + 8*half, n = lane&15) lives in acc[r].
    float* p = partial + ((size_t)bh * NCHUNK + chunk) * PART_STRIDE;
#pragma unroll
    for (int r = 0; r < 8; ++r) {
        const int d = wave * 16 + half * 8 + r;
        p[d * DIM +      l15] = acc0[r];
        p[d * DIM + 16 + l15] = acc1[r];
        p[d * DIM + 32 + l15] = acc2[r];
        p[d * DIM + 48 + l15] = acc3[r];
    }
    if (l15 == 0) {
#pragma unroll
        for (int r = 0; r < 8; ++r)
            p[DIM * DIM + wave * 16 + half * 8 + r] = acc4[r];
    }
}

// ---------------------------------------------------------------------------
// Pass 2: fixed-order reduction of the 8 chunk partials; emit f16 KV^T [e][d]
// with row 64 = k_one, so pass 3's B operand is a contiguous K-major load.
// ---------------------------------------------------------------------------
__global__ __launch_bounds__(256)
void la_pass2_reduce(const float* __restrict__ partial, _Float16* __restrict__ kvt) {
    const int idx = blockIdx.x * 256 + threadIdx.x;
    if (idx >= NBH * KVT_ROWS * DIM) return;
    const int bh  = idx / (KVT_ROWS * DIM);
    const int rem = idx - bh * (KVT_ROWS * DIM);
    const int e   = rem >> 6;
    const int d   = rem & 63;
    const float* p = partial + (size_t)bh * NCHUNK * PART_STRIDE
                   + (e < DIM ? (d * DIM + e) : (DIM * DIM + d));
    float s = 0.0f;
#pragma unroll
    for (int c = 0; c < NCHUNK; ++c) s += p[c * PART_STRIDE];
    kvt[idx] = (_Float16)s;
}

// ---------------------------------------------------------------------------
// Pass 3: out[s,e] = (phi_q[s,:] @ KV[:,e]) / (phi_q[s,:] . k_one + 1e-8).
// Each wave produces a 16(s) x 64(e) tile; normalizer rides as N-tile 4 col 0.
// ---------------------------------------------------------------------------
__global__ __launch_bounds__(128)
void la_pass3_out(const float* __restrict__ Q, const float* __restrict__ mask,
                  const _Float16* __restrict__ kvt, float* __restrict__ out) {
    const int bh   = blockIdx.x >> 6;           // 64 s-groups of 64 rows
    const int sg   = blockIdx.x & 63;
    const int b    = bh >> 4;
    const int lane = threadIdx.x & 31;
    const int wave = threadIdx.x >> 5;
    const int half = (lane >> 4) & 1;
    const int l15  = lane & 15;
    const int sbase = sg * 64 + wave * 16;
    const int srow  = sbase + l15;
    const float scale = 0.35355339059327379f;

    const float* q = Q + ((size_t)bh * SEQ + srow) * DIM;
    const float  m = mask[(size_t)b * SEQ + srow];

    // A = phi_q for this lane's row; per-lane K set is contiguous k = 16*half + s.
    const int b0 = half * 16;
    float ta[16], tb[16];
    *(float4*)(ta + 0)  = *(const float4*)(q + b0 + 0);
    *(float4*)(ta + 4)  = *(const float4*)(q + b0 + 4);
    *(float4*)(ta + 8)  = *(const float4*)(q + b0 + 8);
    *(float4*)(ta + 12) = *(const float4*)(q + b0 + 12);
    *(float4*)(tb + 0)  = *(const float4*)(q + b0 + 32);
    *(float4*)(tb + 4)  = *(const float4*)(q + b0 + 36);
    *(float4*)(tb + 8)  = *(const float4*)(q + b0 + 40);
    *(float4*)(tb + 12) = *(const float4*)(q + b0 + 44);
    v16h a0, a1;
#pragma unroll
    for (int i = 0; i < 16; ++i) {
        a0[i] = (_Float16)(phi_elu1(ta[i] * scale) * m);
        a1[i] = (_Float16)(phi_elu1(tb[i] * scale) * m);
    }

    const _Float16* kb = kvt + (size_t)bh * KVT_ROWS * DIM;
    auto loadB = [&](int nt, int kstep) -> v16h {
        const int e     = nt * 16 + l15;
        const int dbase = kstep * 32 + half * 16;
        return *(const v16h*)(kb + e * DIM + dbase);
    };
    auto loadBnorm = [&](int kstep) -> v16h {   // col 0 = k_one, rest 0
        v16h z = {};
        if (l15 == 0) z = *(const v16h*)(kb + DIM * DIM + kstep * 32 + half * 16);
        return z;
    };

    // Preload all ten B operands into distinct registers (clause-friendly,
    // single wait), then run the WMMA chain back-to-back.
    const v16h b00 = loadB(0, 0), b01 = loadB(0, 1);
    const v16h b10 = loadB(1, 0), b11 = loadB(1, 1);
    const v16h b20 = loadB(2, 0), b21 = loadB(2, 1);
    const v16h b30 = loadB(3, 0), b31 = loadB(3, 1);
    const v16h bn0 = loadBnorm(0), bn1 = loadBnorm(1);

    v8f c0 = {}, c1 = {}, c2 = {}, c3 = {}, c4 = {};
    c0 = wmma16(a0, b00, c0);  c0 = wmma16(a1, b01, c0);
    c1 = wmma16(a0, b10, c1);  c1 = wmma16(a1, b11, c1);
    c2 = wmma16(a0, b20, c2);  c2 = wmma16(a1, b21, c2);
    c3 = wmma16(a0, b30, c3);  c3 = wmma16(a1, b31, c3);
    c4 = wmma16(a0, bn0, c4);  c4 = wmma16(a1, bn1, c4);

    float* orow = out + ((size_t)bh * SEQ + sbase) * DIM;
    const int srcl = half * 16;                 // lane holding column n==0 of this half
#pragma unroll
    for (int r = 0; r < 8; ++r) {
        const float nrm = __shfl(c4[r], srcl, 32) + 1e-8f;
        const float inv = 1.0f / nrm;
        const int   mr  = half * 8 + r;
        orow[mr * DIM +      l15] = c0[r] * inv;
        orow[mr * DIM + 16 + l15] = c1[r] * inv;
        orow[mr * DIM + 32 + l15] = c2[r] * inv;
        orow[mr * DIM + 48 + l15] = c3[r] * inv;
    }
}

extern "C" void kernel_launch(void* const* d_in, const int* in_sizes, int n_in,
                              void* d_out, int out_size, void* d_ws, size_t ws_size,
                              hipStream_t stream) {
    (void)in_sizes; (void)n_in; (void)out_size;
    const float* Q    = (const float*)d_in[0];
    const float* K    = (const float*)d_in[1];
    const float* V    = (const float*)d_in[2];
    const float* mask = (const float*)d_in[3];
    float* out = (float*)d_out;

    // Scratch: prefer d_ws for everything; if it's small, park the chunk
    // partials inside d_out (pass 3 fully overwrites it) and keep only the
    // 520 KB KV^T/k_one buffer in d_ws.
    float*    partial;
    _Float16* kvt;
    if (ws_size >= PART_BYTES + KVT_BYTES) {
        partial = (float*)d_ws;
        kvt     = (_Float16*)((char*)d_ws + PART_BYTES);
    } else {
        partial = (float*)d_out;            // 8.5 MB << 67 MB output buffer
        kvt     = (_Float16*)d_ws;          // needs 520 KB
    }

    la_pass1_kv<<<NBH * NCHUNK, 128, 0, stream>>>(K, V, mask, partial);
    la_pass2_reduce<<<(NBH * KVT_ROWS * DIM + 255) / 256, 256, 0, stream>>>(partial, kvt);
    la_pass3_out<<<NBH * (SEQ / 64), 128, 0, stream>>>(Q, mask, kvt, out);
}
